// GRUAttention_74174085202335
// MI455X (gfx1250) — matrix operand order repaired and checked by reading, not compiled, not run on hardware
//
#include <hip/hip_runtime.h>
#include <hip/hip_bf16.h>

// Problem dims (fixed by the reference)
#define B_  256
#define L_  64
#define DI_ 2048
#define T_  32
#define DT_ 512
#define U_  512
#define A_  512
#define U3_ 1536      // 3*U
#define HC_ 2048      // A + 3U  (fused h-GEMM output width)

typedef __attribute__((ext_vector_type(16))) __bf16       v16bf;
typedef __attribute__((ext_vector_type(8)))  float        v8f;
typedef __attribute__((ext_vector_type(8)))  unsigned int v8u;

__device__ __forceinline__ unsigned short f2bf(float f) {
  unsigned int u = __builtin_bit_cast(unsigned int, f);
  u += 0x7fffu + ((u >> 16) & 1u);          // round-to-nearest-even
  return (unsigned short)(u >> 16);
}
__device__ __forceinline__ unsigned int pack2bf(float lo, float hi) {
  return (unsigned int)f2bf(lo) | ((unsigned int)f2bf(hi) << 16);
}
__device__ __forceinline__ float2 unpack2bf(unsigned int p) {
  return make_float2(__builtin_bit_cast(float, p << 16),
                     __builtin_bit_cast(float, p & 0xffff0000u));
}

// ---------------------------------------------------------------------------
// bf16 WMMA GEMM:  C[M,N](f32) = Abf[M,K](bf16) @ B[K,N] (+bias)
// B pre-transposed+converted: Bt[N,K] bf16.  Block = 256 thr = 8 waves,
// block tile 64(M) x 128(N), wave tile 32x32, 4 accumulators.
// Per K-step: 8x global_load_b128 + 4x v_wmma (no VALU conversion work).
// Requires M%64==0, N%128==0, K%32==0 (all call sites satisfy this).
// ---------------------------------------------------------------------------
__global__ __launch_bounds__(256) void gemm_bf16_wmma(
    const unsigned short* __restrict__ Abf, const unsigned short* __restrict__ Bt,
    const float* __restrict__ bias, float* __restrict__ C, int N, int K)
{
  const int lane = threadIdx.x & 31;
  const int wave = threadIdx.x >> 5;
  const int wm   = wave >> 2;        // 0..1
  const int wn   = wave & 3;         // 0..3
  const int M0   = blockIdx.x * 64 + wm * 32;
  const int N0   = blockIdx.y * 128 + wn * 32;
  const int l15  = lane & 15;
  const int half = lane >> 4;

  const int ar0 = M0 + l15;          // A rows for the two M fragments
  const int ar1 = M0 + 16 + l15;
  const int bc0 = N0 + l15;          // B cols for the two N fragments
  const int bc1 = N0 + 16 + l15;

  // A frag: VGPR v  -> K = 2v..2v+1 (+16*half)
  //   => two contiguous 8-element (16B) runs at +half*16 and +half*16+8.
  // B frag: VGPR v  -> K = 2v..2v+1 (+16*half)
  //   => two contiguous runs at +half*16 and +half*16+8.
  const unsigned short* Arow0 = Abf + (size_t)ar0 * K + half * 16;
  const unsigned short* Arow1 = Abf + (size_t)ar1 * K + half * 16;
  const unsigned short* Brow0 = Bt  + (size_t)bc0 * K + half * 16;
  const unsigned short* Brow1 = Bt  + (size_t)bc1 * K + half * 16;

  v8f acc00 = {}, acc01 = {}, acc10 = {}, acc11 = {};

  for (int kb = 0; kb < K; kb += 32) {
    const uint4 a0lo = *(const uint4*)(Arow0 + kb);
    const uint4 a0hi = *(const uint4*)(Arow0 + kb + 8);
    const uint4 a1lo = *(const uint4*)(Arow1 + kb);
    const uint4 a1hi = *(const uint4*)(Arow1 + kb + 8);
    const uint4 b0lo = *(const uint4*)(Brow0 + kb);
    const uint4 b0hi = *(const uint4*)(Brow0 + kb + 8);
    const uint4 b1lo = *(const uint4*)(Brow1 + kb);
    const uint4 b1hi = *(const uint4*)(Brow1 + kb + 8);

    // Prefetch next K tile (global_prefetch_b8; speculative, safe at edges)
    __builtin_prefetch(Arow0 + kb + 32, 0, 3);
    __builtin_prefetch(Arow1 + kb + 32, 0, 3);
    __builtin_prefetch(Brow0 + kb + 32, 0, 3);
    __builtin_prefetch(Brow1 + kb + 32, 0, 3);

    const v8u ap0 = {a0lo.x, a0lo.y, a0lo.z, a0lo.w, a0hi.x, a0hi.y, a0hi.z, a0hi.w};
    const v8u ap1 = {a1lo.x, a1lo.y, a1lo.z, a1lo.w, a1hi.x, a1hi.y, a1hi.z, a1hi.w};
    const v8u bp0 = {b0lo.x, b0lo.y, b0lo.z, b0lo.w, b0hi.x, b0hi.y, b0hi.z, b0hi.w};
    const v8u bp1 = {b1lo.x, b1lo.y, b1lo.z, b1lo.w, b1hi.x, b1hi.y, b1hi.z, b1hi.w};

    const v16bf a0 = __builtin_bit_cast(v16bf, ap0);
    const v16bf a1 = __builtin_bit_cast(v16bf, ap1);
    const v16bf b0 = __builtin_bit_cast(v16bf, bp0);
    const v16bf b1 = __builtin_bit_cast(v16bf, bp1);
    acc00 = __builtin_amdgcn_wmma_f32_16x16x32_bf16(false, a0, false, b0, (short)0, acc00, false, false);
    acc01 = __builtin_amdgcn_wmma_f32_16x16x32_bf16(false, a0, false, b1, (short)0, acc01, false, false);
    acc10 = __builtin_amdgcn_wmma_f32_16x16x32_bf16(false, a1, false, b0, (short)0, acc10, false, false);
    acc11 = __builtin_amdgcn_wmma_f32_16x16x32_bf16(false, a1, false, b1, (short)0, acc11, false, false);
  }

  // C layout: acc[i] -> row base + i + 8*(lane>=16), col N0 + (lane&15)
  const int   r0 = M0 + (half ? 8 : 0);
  const int   r1 = r0 + 16;
  const float bb0 = bias ? bias[bc0] : 0.f;
  const float bb1 = bias ? bias[bc1] : 0.f;
#pragma unroll
  for (int i = 0; i < 8; ++i) {
    C[(size_t)(r0 + i) * N + bc0] = acc00[i] + bb0;
    C[(size_t)(r0 + i) * N + bc1] = acc01[i] + bb1;
    C[(size_t)(r1 + i) * N + bc0] = acc10[i] + bb0;
    C[(size_t)(r1 + i) * N + bc1] = acc11[i] + bb1;
  }
}

// ---------------------------------------------------------------------------
// Prep kernels (run once per launch)
// ---------------------------------------------------------------------------
// Contiguous fp32 -> bf16 convert, 4 elements/thread
__global__ void cvt_bf16_kernel(const float* __restrict__ in,
                                unsigned short* __restrict__ out)
{
  const int i = (blockIdx.x * blockDim.x + threadIdx.x) * 4;
  const float4 v = *(const float4*)(in + i);
  *(uint2*)(out + i) = make_uint2(pack2bf(v.x, v.y), pack2bf(v.z, v.w));
}

// out[n*K + k] = bf16(in[(k0+k)*Nld + n])   (transpose + convert)
__global__ void transpose_bf16_kernel(const float* __restrict__ in,
                                      unsigned short* __restrict__ out,
                                      int K, int Nld, int k0)
{
  const int idx = blockIdx.x * blockDim.x + threadIdx.x;  // n*K + k
  const int n = idx / K, k = idx - n * K;
  out[idx] = f2bf(in[(size_t)(k0 + k) * Nld + n]);
}

// Fused h-weight: cols 0..511 = att_hidden_kernel, 512..2047 = recurrent_kernel
__global__ void build_wcat_kernel(const float* __restrict__ whid,
                                  const float* __restrict__ wrec,
                                  unsigned short* __restrict__ out)
{
  const int idx = blockIdx.x * blockDim.x + threadIdx.x;  // n*512 + k, n<2048
  const int n = idx >> 9, k = idx & 511;
  const float v = (n < U_) ? whid[(size_t)k * U_ + n]
                           : wrec[(size_t)k * U3_ + (n - U_)];
  out[idx] = f2bf(v);
}

__global__ void build_bcat_kernel(const float* __restrict__ bhid,
                                  const float* __restrict__ brec,
                                  float* __restrict__ out)
{
  const int i = blockIdx.x * blockDim.x + threadIdx.x;    // < 2048
  out[i] = (i < A_) ? bhid[i] : brec[i - A_];
}

__global__ void zero_kernel(unsigned int* __restrict__ p)
{
  p[blockIdx.x * blockDim.x + threadIdx.x] = 0u;
}

// ---------------------------------------------------------------------------
// Attention: score -> softmax(L) -> ctx (bf16).  One block per batch elem b.
// hcat[:, 0:512] = h @ W_att_h + b_att_h (from the fused GEMM).
// ---------------------------------------------------------------------------
__global__ __launch_bounds__(256) void attn_ctx_kernel(
    const float* __restrict__ img_proj,        // [B,L,A] f32
    const float* __restrict__ hcat,            // [B,2048] f32
    const unsigned short* __restrict__ img_bf, // [B,L,DI] bf16
    const float* __restrict__ vk,              // [A]
    const float* __restrict__ vb,              // [1]
    unsigned short* __restrict__ ctx_bf)       // [B,DI] bf16
{
  __shared__ float s_h[A_];
  __shared__ float s_v[A_];
  __shared__ float s_sc[L_];
  const int b = blockIdx.x, tid = threadIdx.x;
  for (int j = tid; j < A_; j += 256) {
    s_h[j] = hcat[(size_t)b * HC_ + j];
    s_v[j] = vk[j];
  }
  __syncthreads();

  const int lane = tid & 31, wave = tid >> 5;
  const float vbias = vb[0];
#pragma unroll
  for (int i = 0; i < 8; ++i) {          // 8 waves x 8 regions = L
    const int l = wave * 8 + i;
    const float* row = img_proj + ((size_t)b * L_ + l) * A_;
    float p = 0.f;
    for (int a = lane; a < A_; a += 32)
      p += tanhf(row[a] + s_h[a]) * s_v[a];
    for (int o = 16; o > 0; o >>= 1) p += __shfl_xor(p, o, 32);
    if (lane == 0) s_sc[l] = p + vbias;
  }
  __syncthreads();

  if (tid < 32) {                        // softmax over L=64 in one wave
    float v0 = s_sc[tid], v1 = s_sc[tid + 32];
    float m = fmaxf(v0, v1);
    for (int o = 16; o > 0; o >>= 1) m = fmaxf(m, __shfl_xor(m, o, 32));
    float e0 = __expf(v0 - m), e1 = __expf(v1 - m);
    float s = e0 + e1;
    for (int o = 16; o > 0; o >>= 1) s += __shfl_xor(s, o, 32);
    const float inv = 1.f / s;
    s_sc[tid] = e0 * inv;
    s_sc[tid + 32] = e1 * inv;
  }
  __syncthreads();

  // ctx[b, d] = sum_l attw[l] * img[b,l,d]; each thread owns 8 contiguous d
  float acc[8] = {0.f, 0.f, 0.f, 0.f, 0.f, 0.f, 0.f, 0.f};
  const int d0 = tid * 8;
#pragma unroll 4
  for (int l = 0; l < L_; ++l) {
    const float w = s_sc[l];
    const uint4 q = *(const uint4*)(img_bf + ((size_t)b * L_ + l) * DI_ + d0);
    const float2 f0 = unpack2bf(q.x), f1 = unpack2bf(q.y);
    const float2 f2 = unpack2bf(q.z), f3 = unpack2bf(q.w);
    acc[0] += w * f0.x; acc[1] += w * f0.y; acc[2] += w * f1.x; acc[3] += w * f1.y;
    acc[4] += w * f2.x; acc[5] += w * f2.y; acc[6] += w * f3.x; acc[7] += w * f3.y;
  }
  *(uint4*)(ctx_bf + (size_t)b * DI_ + d0) =
      make_uint4(pack2bf(acc[0], acc[1]), pack2bf(acc[2], acc[3]),
                 pack2bf(acc[4], acc[5]), pack2bf(acc[6], acc[7]));
}

// ---------------------------------------------------------------------------
// GRU gate math; writes h (f32 + bf16) and out[:,t,:]
// ---------------------------------------------------------------------------
__global__ __launch_bounds__(256) void gate_kernel(
    const float* __restrict__ mx_ctx,   // [B,3U]   ctx @ kernel[DT:]
    const float* __restrict__ mx_text,  // [B,T,3U] text proj (+input_bias)
    const float* __restrict__ hcat,     // [B,2048] cols 512.. = h@Wrec + brec
    float* __restrict__ h,              // [B,U] f32
    unsigned short* __restrict__ h_bf,  // [B,U] bf16 (GEMM A operand)
    float* __restrict__ out,            // [B,T,U]
    int t)
{
  const int idx = blockIdx.x * 256 + threadIdx.x;   // < B*U
  const int b = idx >> 9, u = idx & 511;
  const float* mx = mx_ctx + (size_t)b * U3_;
  const float* mt = mx_text + ((size_t)b * T_ + t) * U3_;
  const float* mi = hcat + (size_t)b * HC_ + U_;
  const float x_z = mx[u]          + mt[u];
  const float x_r = mx[u + U_]     + mt[u + U_];
  const float x_h = mx[u + 2 * U_] + mt[u + 2 * U_];
  const float r_z = mi[u], r_r = mi[u + U_], r_h = mi[u + 2 * U_];
  const float z  = 1.f / (1.f + __expf(-(x_z + r_z)));
  const float r  = 1.f / (1.f + __expf(-(x_r + r_r)));
  const float hh = tanhf(x_h + r * r_h);
  const float hn = z * h[idx] + (1.f - z) * hh;
  h[idx]    = hn;
  h_bf[idx] = f2bf(hn);
  out[((size_t)b * T_ + t) * U_ + u] = hn;
}

// ---------------------------------------------------------------------------
extern "C" void kernel_launch(void* const* d_in, const int* in_sizes, int n_in,
                              void* d_out, int out_size, void* d_ws, size_t ws_size,
                              hipStream_t stream)
{
  const float* img      = (const float*)d_in[0];   // [B,L,DI]
  const float* text     = (const float*)d_in[1];   // [B,T,DT]
  const float* kernel_  = (const float*)d_in[2];   // [DT+DI, 3U]
  const float* in_bias  = (const float*)d_in[3];   // [3U]
  const float* rec_k    = (const float*)d_in[4];   // [U, 3U]
  const float* rec_b    = (const float*)d_in[5];   // [3U]
  const float* w_ai     = (const float*)d_in[6];   // [DI, A]
  const float* b_ai     = (const float*)d_in[7];   // [A]
  const float* w_ah     = (const float*)d_in[8];   // [U, A]
  const float* b_ah     = (const float*)d_in[9];   // [A]
  const float* v_k      = (const float*)d_in[10];  // [A,1]
  const float* v_b      = (const float*)d_in[11];  // [1]
  float* out = (float*)d_out;

  // ---- workspace carve-up -------------------------------------------------
  char* w = (char*)d_ws;
  size_t off = 0;
  auto carve = [&](size_t bytes) -> void* {
    void* p = w + off;
    off += (bytes + 255) & ~(size_t)255;
    return p;
  };
  float* img_proj = (float*)carve((size_t)B_ * L_ * A_ * 4);     // 33.5 MB
  float* mx_text  = (float*)carve((size_t)B_ * T_ * U3_ * 4);    // 50.3 MB
  float* h        = (float*)carve((size_t)B_ * U_ * 4);
  float* hcat     = (float*)carve((size_t)B_ * HC_ * 4);
  float* mx_ctx   = (float*)carve((size_t)B_ * U3_ * 4);
  float* bcat     = (float*)carve((size_t)HC_ * 4);
  unsigned short* img_bf  = (unsigned short*)carve((size_t)B_ * L_ * DI_ * 2); // 67 MB
  unsigned short* text_bf = (unsigned short*)carve((size_t)B_ * T_ * DT_ * 2); // 8.4 MB
  unsigned short* h_bf    = (unsigned short*)carve((size_t)B_ * U_ * 2);
  unsigned short* ctx_bf  = (unsigned short*)carve((size_t)B_ * DI_ * 2);
  unsigned short* Wt_ai   = (unsigned short*)carve((size_t)A_  * DI_ * 2);
  unsigned short* Wt_txt  = (unsigned short*)carve((size_t)U3_ * DT_ * 2);
  unsigned short* Wt_ctx  = (unsigned short*)carve((size_t)U3_ * DI_ * 2);
  unsigned short* Wt_cat  = (unsigned short*)carve((size_t)HC_ * U_  * 2);
  (void)ws_size; (void)in_sizes; (void)n_in; (void)out_size;

  // ---- one-time prep ------------------------------------------------------
  cvt_bf16_kernel<<<(B_ * L_ * DI_) / 1024, 256, 0, stream>>>(img, img_bf);
  cvt_bf16_kernel<<<(B_ * T_ * DT_) / 1024, 256, 0, stream>>>(text, text_bf);
  transpose_bf16_kernel<<<(A_ * DI_) / 256, 256, 0, stream>>>(w_ai, Wt_ai, DI_, A_, 0);
  transpose_bf16_kernel<<<(U3_ * DT_) / 256, 256, 0, stream>>>(kernel_, Wt_txt, DT_, U3_, 0);
  transpose_bf16_kernel<<<(U3_ * DI_) / 256, 256, 0, stream>>>(kernel_, Wt_ctx, DI_, U3_, DT_);
  build_wcat_kernel<<<(HC_ * U_) / 256, 256, 0, stream>>>(w_ah, rec_k, Wt_cat);
  build_bcat_kernel<<<HC_ / 256, 256, 0, stream>>>(b_ah, rec_b, bcat);
  zero_kernel<<<(B_ * U_) / 256, 256, 0, stream>>>((unsigned int*)h);
  zero_kernel<<<(B_ * U_) / 512, 256, 0, stream>>>((unsigned int*)h_bf);

  // img_proj = img @ W_ai + b_ai           (M=16384, N=512, K=2048)
  gemm_bf16_wmma<<<dim3((B_ * L_) / 64, A_ / 128), 256, 0, stream>>>(
      img_bf, Wt_ai, b_ai, img_proj, A_, DI_);
  // mx_text = text @ kernel[:DT] + in_bias (M=8192, N=1536, K=512)
  gemm_bf16_wmma<<<dim3((B_ * T_) / 64, U3_ / 128), 256, 0, stream>>>(
      text_bf, Wt_txt, in_bias, mx_text, U3_, DT_);

  // ---- sequential scan over T timesteps ----------------------------------
  for (int t = 0; t < T_; ++t) {
    // hcat = h @ [W_att_h | W_rec] + [b_att_h | b_rec]  (M=256, N=2048, K=512)
    gemm_bf16_wmma<<<dim3(B_ / 64, HC_ / 128), 256, 0, stream>>>(
        h_bf, Wt_cat, bcat, hcat, HC_, U_);
    // Bahdanau attention + context vector (bf16 out)
    attn_ctx_kernel<<<B_, 256, 0, stream>>>(img_proj, hcat, img_bf, v_k, v_b, ctx_bf);
    // mx_ctx = ctx @ kernel[DT:]             (M=256, N=1536, K=2048)
    gemm_bf16_wmma<<<dim3(B_ / 64, U3_ / 128), 256, 0, stream>>>(
        ctx_bf, Wt_ctx, (const float*)nullptr, mx_ctx, U3_, DI_);
    // GRU gates -> h, h_bf, out[:, t, :]
    gate_kernel<<<(B_ * U_) / 256, 256, 0, stream>>>(
        mx_ctx, mx_text, hcat, h, h_bf, out, t);
  }
}
